// CrossLayerTranscoder_32830730011583
// MI455X (gfx1250) — matrix operand order repaired and checked by reading, not compiled, not run on hardware
//
#include <hip/hip_runtime.h>
#include <hip/hip_bf16.h>

// CrossLayerTranscoder for MI455X (gfx1250, wave32, WMMA).
// HBM-bound (~2.5 GB mandatory traffic vs ~2.3 TFLOP): bf16 WMMA with
// on-the-fly fp32->bf16 conversion, register-staged double-buffered LDS
// pipeline (1 barrier / k-step), decoder accumulates the 78 (i,l) pair GEMMs
// into L2-resident logits via fp32 global atomics.

typedef __attribute__((ext_vector_type(16))) __bf16 v16bf;
typedef __attribute__((ext_vector_type(8)))  __bf16 v8bf;
typedef __attribute__((ext_vector_type(8)))  float  v8f;

#define B_DIM 4096
#define L_DIM 12
#define A_DIM 768
#define F_DIM 4096

#define BM 128
#define BN 128
#define BK 32
#define LDT 40  // padded LDS row stride (bf16 elems): 80B rows, 16B-aligned

static_assert(B_DIM % BM == 0, "");
static_assert(F_DIM % BN == 0 && F_DIM % BK == 0, "");
static_assert(A_DIM % BN == 0 && A_DIM % BK == 0, "");

// ---------------------------------------------------------------------------
// Fragment helpers (CDNA5 ISA 7.12.2 VGPR layouts, wave32)
// A 16x32 bf16: lane m<16 -> row m, K {0..7},{16..23}; lane m+16 -> K {8..15},{24..31}
// B 32x16 bf16: lane n<16 -> col n, K {0..7},{16..23}; lane n+16 -> K {8..15},{24..31}
// C/D 16x16 f32: lane n<16 -> col n, M = vgpr idx; lane n+16 -> M = 8 + vgpr idx
// ---------------------------------------------------------------------------
__device__ __forceinline__ v16bf load_a_frag(const __bf16* row_base, int half) {
  v16bf a;
  const __bf16* p = row_base + half * 8;
  *(v8bf*)&a       = *(const v8bf*)(p);
  *((v8bf*)&a + 1) = *(const v8bf*)(p + 16);
  return a;
}
__device__ __forceinline__ v16bf load_b_frag(const __bf16* row_base, int half) {
  v16bf b;
  const __bf16* p = row_base + half * 8;
  *(v8bf*)&b       = *(const v8bf*)(p);
  *((v8bf*)&b + 1) = *(const v8bf*)(p + 16);
  return b;
}

__device__ __forceinline__ void mma_step(const __bf16* As, const __bf16* Bs,
                                         v8f (&acc)[2][4],
                                         int wm, int wn, int n16, int half) {
  v16bf af[2], bf[4];
#pragma unroll
  for (int mi = 0; mi < 2; ++mi)
    af[mi] = load_a_frag(As + (wm + mi * 16 + n16) * LDT, half);
#pragma unroll
  for (int ni = 0; ni < 4; ++ni)
    bf[ni] = load_b_frag(Bs + (wn + ni * 16 + n16) * LDT, half);
#pragma unroll
  for (int mi = 0; mi < 2; ++mi)
#pragma unroll
    for (int ni = 0; ni < 4; ++ni)
      acc[mi][ni] = __builtin_amdgcn_wmma_f32_16x16x32_bf16(
          false, af[mi], false, bf[ni], (short)0, acc[mi][ni], false, false);
}

struct Stage {
  float4 a[4];    // A-tile data (4 rows of 4 floats, same k-slice)
  float4 b[4];    // B-tile data (4 k-rows of 4 cols)
  float4 s0, s1;  // per-k scale/threshold operands
};

// ===========================================================================
// Encoder: enc[b,l,f] = sum_a ((x[b,l,a]-mean[l,a])/std[l,a]) * W_enc[l,a,f]
// grid: (x = b-tile [innermost: W_enc slab L2 reuse], y = f-tile, z = l)
// ===========================================================================
__global__ __launch_bounds__(256) void clt_encoder(
    const float* __restrict__ x, const float* __restrict__ in_mean,
    const float* __restrict__ in_std, const float* __restrict__ W_enc,
    float* __restrict__ enc_out)
{
  __shared__ __align__(16) __bf16 As[2][BM * LDT];  // [m(b)][k(a)]
  __shared__ __align__(16) __bf16 Bs[2][BN * LDT];  // transposed [n(f)][k(a)]

  const int tid  = threadIdx.x;
  const int b0   = blockIdx.x * BM;
  const int f0   = blockIdx.y * BN;
  const int l    = blockIdx.z;
  const int lane = tid & 31;
  const int wv   = tid >> 5;
  const int wm   = (wv & 3) * 32;
  const int wn   = (wv >> 2) * 64;
  const int n16  = lane & 15;
  const int half = lane >> 4;

  const int kA = (tid & 7) * 4, rowA = tid >> 3;  // A loader slice
  const int cB = (tid & 31) * 4, rB = tid >> 5;   // B loader slice

  auto load = [&](int k0, Stage& st) {
    st.s0 = *(const float4*)(in_mean + (size_t)l * A_DIM + k0 + kA);
    st.s1 = *(const float4*)(in_std  + (size_t)l * A_DIM + k0 + kA);
#pragma unroll
    for (int p = 0; p < 4; ++p)
      st.a[p] = *(const float4*)(x + ((size_t)(b0 + rowA + 32 * p) * L_DIM + l) * A_DIM + k0 + kA);
#pragma unroll
    for (int p = 0; p < 4; ++p)
      st.b[p] = *(const float4*)(W_enc + ((size_t)l * A_DIM + k0 + rB + 8 * p) * F_DIM + f0 + cB);
  };
  auto stage_to_lds = [&](__bf16* as, __bf16* bs, const Stage& st) {
    const float rx = __builtin_amdgcn_rcpf(st.s1.x);
    const float ry = __builtin_amdgcn_rcpf(st.s1.y);
    const float rz = __builtin_amdgcn_rcpf(st.s1.z);
    const float rw = __builtin_amdgcn_rcpf(st.s1.w);
#pragma unroll
    for (int p = 0; p < 4; ++p) {
      union { __bf16 h[4]; uint2 u; } pk;
      pk.h[0] = (__bf16)((st.a[p].x - st.s0.x) * rx);
      pk.h[1] = (__bf16)((st.a[p].y - st.s0.y) * ry);
      pk.h[2] = (__bf16)((st.a[p].z - st.s0.z) * rz);
      pk.h[3] = (__bf16)((st.a[p].w - st.s0.w) * rw);
      *(uint2*)(as + (rowA + 32 * p) * LDT + kA) = pk.u;
    }
#pragma unroll
    for (int p = 0; p < 4; ++p) {
      bs[(cB + 0) * LDT + rB + 8 * p] = (__bf16)st.b[p].x;
      bs[(cB + 1) * LDT + rB + 8 * p] = (__bf16)st.b[p].y;
      bs[(cB + 2) * LDT + rB + 8 * p] = (__bf16)st.b[p].z;
      bs[(cB + 3) * LDT + rB + 8 * p] = (__bf16)st.b[p].w;
    }
  };

  v8f acc[2][4] = {};
  Stage st;
  load(0, st);
  stage_to_lds(As[0], Bs[0], st);
  __syncthreads();
  int cur = 0;
  for (int k0 = BK; k0 < A_DIM; k0 += BK) {
    load(k0, st);                                  // issue next tile's loads
    mma_step(As[cur], Bs[cur], acc, wm, wn, n16, half);  // hide latency in WMMA
    stage_to_lds(As[cur ^ 1], Bs[cur ^ 1], st);
    __syncthreads();
    cur ^= 1;
  }
  mma_step(As[cur], Bs[cur], acc, wm, wn, n16, half);

  // write pre-threshold enc (fp32)
#pragma unroll
  for (int mi = 0; mi < 2; ++mi)
#pragma unroll
    for (int ni = 0; ni < 4; ++ni)
#pragma unroll
      for (int r = 0; r < 8; ++r) {
        int row = b0 + wm + mi * 16 + half * 8 + r;
        int col = f0 + wn + ni * 16 + n16;
        enc_out[((size_t)row * L_DIM + l) * F_DIM + col] = acc[mi][ni][r];
      }
}

// ===========================================================================
// Decoder: for pair (i,l), l<=i:  logits[:, i, :] += act_l @ W_dec[tri(i,l)]
//   act_l[b,f] = enc[b,l,f] * (enc > theta[l,f])   (fused JumpReLU)
// grid: (x = b-tile [innermost: W_dec slab reuse], y = a-tile,
//        z = pair index ordered l-major so act slice l stays L2-resident)
// ===========================================================================
__global__ __launch_bounds__(256) void clt_decoder(
    const float* __restrict__ enc_in, const float* __restrict__ theta,
    const float* __restrict__ W_dec, float* __restrict__ logits)
{
  __shared__ __align__(16) __bf16 As[2][BM * LDT];  // [m(b)][k(f)]
  __shared__ __align__(16) __bf16 Bs[2][BN * LDT];  // transposed [n(a)][k(f)]

  // pair index p (l-major) -> (l, i); W_dec_packed index t = i(i+1)/2 + l
  int p = blockIdx.z;
  int l = 0;
  while (p >= (L_DIM - l)) { p -= (L_DIM - l); ++l; }
  const int li = l + p;
  const int t  = li * (li + 1) / 2 + l;

  const int tid  = threadIdx.x;
  const int b0   = blockIdx.x * BM;
  const int a0   = blockIdx.y * BN;
  const int lane = tid & 31;
  const int wv   = tid >> 5;
  const int wm   = (wv & 3) * 32;
  const int wn   = (wv >> 2) * 64;
  const int n16  = lane & 15;
  const int half = lane >> 4;

  const int kA = (tid & 7) * 4, rowA = tid >> 3;
  const int cB = (tid & 31) * 4, rB = tid >> 5;

  auto load = [&](int k0, Stage& st) {
    st.s0 = *(const float4*)(theta + (size_t)l * F_DIM + k0 + kA);
#pragma unroll
    for (int p2 = 0; p2 < 4; ++p2)
      st.a[p2] = *(const float4*)(enc_in + ((size_t)(b0 + rowA + 32 * p2) * L_DIM + l) * F_DIM + k0 + kA);
#pragma unroll
    for (int p2 = 0; p2 < 4; ++p2)
      st.b[p2] = *(const float4*)(W_dec + ((size_t)t * F_DIM + k0 + rB + 8 * p2) * A_DIM + a0 + cB);
  };
  auto stage_to_lds = [&](__bf16* as, __bf16* bs, const Stage& st) {
#pragma unroll
    for (int p2 = 0; p2 < 4; ++p2) {
      union { __bf16 h[4]; uint2 u; } pk;
      pk.h[0] = (__bf16)(st.a[p2].x > st.s0.x ? st.a[p2].x : 0.0f);
      pk.h[1] = (__bf16)(st.a[p2].y > st.s0.y ? st.a[p2].y : 0.0f);
      pk.h[2] = (__bf16)(st.a[p2].z > st.s0.z ? st.a[p2].z : 0.0f);
      pk.h[3] = (__bf16)(st.a[p2].w > st.s0.w ? st.a[p2].w : 0.0f);
      *(uint2*)(as + (rowA + 32 * p2) * LDT + kA) = pk.u;
    }
#pragma unroll
    for (int p2 = 0; p2 < 4; ++p2) {
      bs[(cB + 0) * LDT + rB + 8 * p2] = (__bf16)st.b[p2].x;
      bs[(cB + 1) * LDT + rB + 8 * p2] = (__bf16)st.b[p2].y;
      bs[(cB + 2) * LDT + rB + 8 * p2] = (__bf16)st.b[p2].z;
      bs[(cB + 3) * LDT + rB + 8 * p2] = (__bf16)st.b[p2].w;
    }
  };

  v8f acc[2][4] = {};
  Stage st;
  load(0, st);
  stage_to_lds(As[0], Bs[0], st);
  __syncthreads();
  int cur = 0;
  for (int k0 = BK; k0 < F_DIM; k0 += BK) {
    load(k0, st);
    mma_step(As[cur], Bs[cur], acc, wm, wn, n16, half);
    stage_to_lds(As[cur ^ 1], Bs[cur ^ 1], st);
    __syncthreads();
    cur ^= 1;
  }
  mma_step(As[cur], Bs[cur], acc, wm, wn, n16, half);

  // accumulate into L2-resident logits
#pragma unroll
  for (int mi = 0; mi < 2; ++mi)
#pragma unroll
    for (int ni = 0; ni < 4; ++ni)
#pragma unroll
      for (int r = 0; r < 8; ++r) {
        int row = b0 + wm + mi * 16 + half * 8 + r;
        int col = a0 + wn + ni * 16 + n16;
        atomicAdd(logits + ((size_t)row * L_DIM + li) * A_DIM + col, acc[mi][ni][r]);
      }
}

// ===========================================================================
extern "C" void kernel_launch(void* const* d_in, const int* in_sizes, int n_in,
                              void* d_out, int out_size, void* d_ws, size_t ws_size,
                              hipStream_t stream) {
  const float* x       = (const float*)d_in[0];
  const float* in_mean = (const float*)d_in[1];
  const float* in_std  = (const float*)d_in[2];
  const float* W_enc   = (const float*)d_in[3];
  const float* theta   = (const float*)d_in[4];
  const float* W_dec   = (const float*)d_in[5];

  float* logits = (float*)d_out;                          // (B, L, A)
  float* enc    = logits + (size_t)B_DIM * L_DIM * A_DIM; // (B, L, F)

  // logits accumulated via atomics -> must be zeroed every call
  hipMemsetAsync(logits, 0, sizeof(float) * (size_t)B_DIM * L_DIM * A_DIM, stream);

  dim3 blk(256);
  dim3 g_enc(B_DIM / BM, F_DIM / BN, L_DIM);
  clt_encoder<<<g_enc, blk, 0, stream>>>(x, in_mean, in_std, W_enc, enc);

  dim3 g_dec(B_DIM / BM, A_DIM / BN, L_DIM * (L_DIM + 1) / 2);
  clt_decoder<<<g_dec, blk, 0, stream>>>(enc, theta, W_dec, logits);
}
